// NsaAttention_42812234007032
// MI455X (gfx1250) — matrix operand order repaired and checked by reading, not compiled, not run on hardware
//
#include <hip/hip_runtime.h>
#include <hip/hip_bf16.h>
#include <hip/hip_fp16.h>

typedef __attribute__((ext_vector_type(16))) _Float16 v16h;
typedef __attribute__((ext_vector_type(8)))  _Float16 v8h;
typedef __attribute__((ext_vector_type(8)))  float    v8f;

#define KSZ     32
#define STRIDE_ 16
#define N_      8192
#define QH_     32
#define KH_     2
#define G_      16
#define D_      128
#define M_      511      // (8192-32)/16 + 1
#define MPAD    512
#define NEGINF  (-1e30f)
#define SM_SCALE 0.08838834764831845f  // 1/sqrt(128)

// ---------------------------------------------------------------------------
// Kernel 1: block-compress k and v into f16 workspace.
//   ckh: [KH][MPAD][D]   (d contiguous)  -- A-operand friendly
//   cvT: [KH][D][MPAD]   (m contiguous)  -- B-operand friendly (pre-transposed)
// Pad block m=511 is written as zeros.
// ---------------------------------------------------------------------------
__global__ __launch_bounds__(256) void nsa_compress_kernel(
    const float* __restrict__ k,
    const float* __restrict__ v,
    const float* __restrict__ w,
    const float* __restrict__ pe,
    _Float16* __restrict__ ckh,
    _Float16* __restrict__ cvT)
{
    const int tid = blockIdx.x * blockDim.x + threadIdx.x; // KH*MPAD*D threads
    const int d  = tid & (D_ - 1);
    const int m  = (tid >> 7) & (MPAD - 1);
    const int kh = tid >> 16;
    if (kh >= KH_) return;

    float wsum = 0.f;
#pragma unroll
    for (int t = 0; t < KSZ; ++t) wsum += w[t];
    wsum = fmaxf(wsum, 1e-6f);
    const float inv_wsum = 1.f / wsum;

    float ak = 0.f, av = 0.f;
    if (m < M_) {
        const int base = (m * STRIDE_) * (KH_ * D_) + kh * D_ + d;
#pragma unroll 4
        for (int t = 0; t < KSZ; ++t) {
            const float wt = w[t];
            const float p  = pe[t * D_ + d];
            ak = fmaf(k[base + t * (KH_ * D_)] + p, wt, ak);
            av = fmaf(v[base + t * (KH_ * D_)] + p, wt, av);
        }
        ak *= inv_wsum;
        av *= inv_wsum;
    }
    ckh[((size_t)kh * MPAD + m) * D_ + d] = (_Float16)ak;
    cvT[((size_t)kh * D_ + d) * MPAD + m] = (_Float16)av;
}

// ---------------------------------------------------------------------------
// Kernel 2: one wave per (n, kh) tile. WMMA tile rows = the 16 query heads in
// the GQA group (uniform causal mask per tile). Computes S^T = CK * Q^T so the
// softmaxed P converts to the PV A-operand with a pure per-lane f32->f16 pack.
// Chunk loop is software-pipelined: all A-frag loads issued up front; PV
// B-frag loads issued before the softmax VALU phase so their latency hides
// behind exp/shfl work.
// ---------------------------------------------------------------------------
__global__ __launch_bounds__(256) void nsa_attn_kernel(
    const float*    __restrict__ q,
    const _Float16* __restrict__ ckh,
    const _Float16* __restrict__ cvT,
    float*          __restrict__ out)
{
    const int lane   = threadIdx.x & 31;
    const int wave   = threadIdx.x >> 5;
    const int tile   = blockIdx.x * 8 + wave;     // 0 .. N*KH-1
    const int n      = tile >> 1;
    const int kh     = tile & 1;
    const int laneHi = lane >> 4;                 // 0 or 1
    const int lmod   = lane & 15;

    const int Mv = (n >= KSZ - 1) ? ((n - (KSZ - 1)) / STRIDE_ + 1) : 0;

    float* op = out + ((size_t)n * QH_ + kh * G_) * D_;

    if (Mv == 0) {  // no visible block: output zeros for these 16 heads
        for (int i = lane; i < G_ * D_; i += 32) op[i] = 0.f;
        return;
    }

    // --- Load Q fragments (B operand of S^T gemm), K=32 per chunk, 4 chunks.
    // B layout (32x16 K x N, 16-bit): lane column = lane&15 (head g),
    // halves[0..7] = K = dbase + laneHi*8 + 0..7, halves[8..15] = K = dbase + 16 + laneHi*8 + 0..7.
    v16h qf[4];
    {
        const float* qb = q + (((size_t)n * QH_ + kh * G_ + lmod) * D_) + laneHi * 8;
#pragma unroll
        for (int c = 0; c < 4; ++c) {
            v8f lo = *(const v8f*)(qb + c * 32);
            v8f hi = *(const v8f*)(qb + c * 32 + 16);
#pragma unroll
            for (int i = 0; i < 8; ++i) {
                qf[c][i]     = (_Float16)lo[i];
                qf[c][8 + i] = (_Float16)hi[i];
            }
        }
    }

    const _Float16* ckb = ckh + (size_t)kh * MPAD * D_;
    const _Float16* cvb = cvT + (size_t)kh * D_ * MPAD;

    float m_run = NEGINF;
    float l_run = 0.f;
    v8f oacc[8];
#pragma unroll
    for (int j = 0; j < 8; ++j) oacc[j] = (v8f){};

    for (int m0 = 0; m0 < Mv; m0 += 32) {
        // ---- Phase 1: issue ALL A-frag loads for this chunk (16 x b128),
        // then run the 8 score WMMAs. af[2c] = blocks m0..+15, af[2c+1] = +16..31.
        v16h af[8];
        {
            const _Float16* arow = ckb + (size_t)(m0 + lmod) * D_ + laneHi * 8;
#pragma unroll
            for (int c = 0; c < 4; ++c) {
                const _Float16* ap0 = arow + c * 32;
                const _Float16* ap1 = ap0 + 16 * D_;
                v8h a0lo = *(const v8h*)ap0;
                v8h a0hi = *(const v8h*)(ap0 + 16);
                v8h a1lo = *(const v8h*)ap1;
                v8h a1hi = *(const v8h*)(ap1 + 16);
#pragma unroll
                for (int i = 0; i < 8; ++i) {
                    af[2 * c][i]         = a0lo[i];
                    af[2 * c][8 + i]     = a0hi[i];
                    af[2 * c + 1][i]     = a1lo[i];
                    af[2 * c + 1][8 + i] = a1hi[i];
                }
            }
        }

        v8f s0 = (v8f){};
        v8f s1 = (v8f){};
#pragma unroll
        for (int c = 0; c < 4; ++c) {
            s0 = __builtin_amdgcn_wmma_f32_16x16x32_f16(false, af[2 * c],     false, qf[c],
                                                        (short)0, s0, false, false);
            s1 = __builtin_amdgcn_wmma_f32_16x16x32_f16(false, af[2 * c + 1], false, qf[c],
                                                        (short)0, s1, false, false);
        }

        // ---- Phase 2: issue the PV B-frag loads NOW (16 x b128); their
        // latency hides behind the whole softmax VALU phase below.
        // B layout: lane column d = j*16 + lane&15,
        // halves[0..7] = m = m0 + laneHi*8 + 0..7, halves[8..15] = m = m0 + 16 + laneHi*8 + 0..7.
        v16h bf[8];
        {
            const _Float16* bcol = cvb + (size_t)lmod * MPAD + m0 + laneHi * 8;
#pragma unroll
            for (int j = 0; j < 8; ++j) {
                const _Float16* bp = bcol + (size_t)(j * 16) * MPAD;
                v8h blo = *(const v8h*)bp;
                v8h bhi = *(const v8h*)(bp + 16);
#pragma unroll
                for (int i = 0; i < 8; ++i) { bf[j][i] = blo[i]; bf[j][8 + i] = bhi[i]; }
            }
        }

        // ---- scale + mask + online softmax (row = head = lane&15 column of S^T)
        const int lim = Mv - m0;
        float cmax = NEGINF;
#pragma unroll
        for (int r = 0; r < 8; ++r) {
            const int mrel0 = r + 8 * laneHi;        // block offset of s0[r]
            const int mrel1 = mrel0 + 16;            // block offset of s1[r]
            float x0 = (mrel0 < lim) ? s0[r] * SM_SCALE : NEGINF;
            float x1 = (mrel1 < lim) ? s1[r] * SM_SCALE : NEGINF;
            s0[r] = x0; s1[r] = x1;
            cmax = fmaxf(cmax, fmaxf(x0, x1));
        }
        cmax = fmaxf(cmax, __shfl_xor(cmax, 16, 32));
        const float m_new = fmaxf(m_run, cmax);
        const float alpha = __expf(m_run - m_new);

        // ---- P pack: C-layout (vgpr=block, lane=head) -> A-layout per-lane pack
        v16h pf;
        float csum = 0.f;
#pragma unroll
        for (int r = 0; r < 8; ++r) {
            float p0 = __expf(s0[r] - m_new);
            float p1 = __expf(s1[r] - m_new);
            csum += p0 + p1;
            pf[r]     = (_Float16)p0;   // halves[0..7]  = blocks {0..7}/{8..15}
            pf[8 + r] = (_Float16)p1;   // halves[8..15] = blocks {16..23}/{24..31}
        }
        csum += __shfl_xor(csum, 16, 32);
        l_run = l_run * alpha + csum;
        m_run = m_new;

        // ---- rescale accumulator rows by alpha of their head
        float arow[8];
#pragma unroll
        for (int r = 0; r < 8; ++r) arow[r] = __shfl(alpha, r + 8 * laneHi, 32);
#pragma unroll
        for (int j = 0; j < 8; ++j) {
#pragma unroll
            for (int r = 0; r < 8; ++r) oacc[j][r] *= arow[r];
        }

        // ---- PV: O += P(16g x 32m) * CV(32m x 16d), 8 d-column tiles
#pragma unroll
        for (int j = 0; j < 8; ++j) {
            oacc[j] = __builtin_amdgcn_wmma_f32_16x16x32_f16(false, pf, false, bf[j],
                                                             (short)0, oacc[j],
                                                             false, false);
        }
    }

    // ---- epilogue: divide by row sums and store
    float inv[8];
#pragma unroll
    for (int r = 0; r < 8; ++r) {
        float lr = __shfl(l_run, r + 8 * laneHi, 32);
        inv[r] = (lr > 0.f) ? (1.f / lr) : 0.f;
    }
#pragma unroll
    for (int j = 0; j < 8; ++j) {
#pragma unroll
        for (int r = 0; r < 8; ++r) {
            const int g = r + 8 * laneHi;
            const int d = j * 16 + lmod;
            op[g * D_ + d] = oacc[j][r] * inv[r];
        }
    }
}

// ---------------------------------------------------------------------------
extern "C" void kernel_launch(void* const* d_in, const int* in_sizes, int n_in,
                              void* d_out, int out_size, void* d_ws, size_t ws_size,
                              hipStream_t stream) {
    const float* q  = (const float*)d_in[0];
    const float* k  = (const float*)d_in[1];
    const float* v  = (const float*)d_in[2];
    const float* w  = (const float*)d_in[3];
    const float* pe = (const float*)d_in[4];
    float* out = (float*)d_out;

    // workspace: ckh [KH][MPAD][D] f16 (256 KB), cvT [KH][D][MPAD] f16 (256 KB)
    _Float16* ckh = (_Float16*)d_ws;
    _Float16* cvT = ckh + (size_t)KH_ * MPAD * D_;

    {   // compression: KH*MPAD*D = 131072 threads
        const int total = KH_ * MPAD * D_;
        nsa_compress_kernel<<<total / 256, 256, 0, stream>>>(k, v, w, pe, ckh, cvT);
    }
    {   // attention: one wave per (n, kh) tile; 8 waves per block
        const int tiles  = N_ * KH_;         // 16384
        const int blocks = tiles / 8;        // 2048
        nsa_attn_kernel<<<blocks, 256, 0, stream>>>(q, ckh, cvT, out);
    }
}
